// Sparse_Three_Concat_41343355191555
// MI455X (gfx1250) — matrix operand order repaired and checked by reading, not compiled, not run on hardware
//
#include <hip/hip_runtime.h>
#include <hip/hip_bf16.h>

#define NNODES 100000
#define NEDGES 1600000
#define FIN 256
#define HID 128
#define CLS 64
#define FCAT 512

typedef __attribute__((ext_vector_type(16))) __bf16 v16bf;
typedef __attribute__((ext_vector_type(8)))  float  v8f;

union ABu { v16bf v; uint4 q[2]; };

__device__ __forceinline__ unsigned short f2bf(float f) {
    unsigned int u = __float_as_uint(f);
    u += 0x7FFFu + ((u >> 16) & 1u);   // round-to-nearest-even
    return (unsigned short)(u >> 16);
}

// ---------------- conversion / packing ----------------

__global__ void k_cvt_bf16(const float* __restrict__ in, unsigned short* __restrict__ out, long long n) {
    long long i = (long long)blockIdx.x * blockDim.x + threadIdx.x;
    if (i < n) out[i] = f2bf(in[i]);
}

// Pack row-major W[K x Ncols] into WMMA-B tile layout:
// tile (nt,kt): lane l holds col n = nt*16+(l&15), K-pairs kk = kt*32 + (l>>4)*16 + 2v (+which)
// stored contiguously: Bp[((nt*KT+kt)*32 + l)*16 + 2v + which]
__global__ void k_pack_b(const float* __restrict__ W, unsigned short* __restrict__ Bp, int K, int Ncols) {
    int i = blockIdx.x * blockDim.x + threadIdx.x;
    if (i >= K * Ncols) return;
    int which = i & 1;
    int pairi = i >> 1;
    int v     = pairi & 7;
    int lane  = (pairi >> 3) & 31;
    int tile  = pairi >> 8;
    int KT = K >> 5;
    int kt = tile % KT, nt = tile / KT;
    int nn = nt * 16 + (lane & 15);
    int kk = kt * 32 + ((lane >> 4) * 16) + 2 * v + which;
    Bp[i] = f2bf(W[(size_t)kk * Ncols + nn]);
}

// Pack virtual Wcat[256 x 512] = [W_ln | W1 | W2 | W3] (each 256x128 row-major)
__global__ void k_pack_wcat(const float* __restrict__ Wln, const float* __restrict__ W1,
                            const float* __restrict__ W2,  const float* __restrict__ W3,
                            unsigned short* __restrict__ Bp) {
    int i = blockIdx.x * blockDim.x + threadIdx.x;
    if (i >= FIN * FCAT) return;
    int which = i & 1;
    int pairi = i >> 1;
    int v     = pairi & 7;
    int lane  = (pairi >> 3) & 31;
    int tile  = pairi >> 8;
    const int KT = FIN >> 5;       // 8
    int kt = tile % KT, nt = tile / KT;
    int nn = nt * 16 + (lane & 15);
    int kk = kt * 32 + ((lane >> 4) * 16) + 2 * v + which;
    const float* Ws = (nn < 128) ? Wln : (nn < 256) ? W1 : (nn < 384) ? W2 : W3;
    Bp[i] = f2bf(Ws[(size_t)kk * HID + (nn & 127)]);
}

// ---------------- degrees ----------------

__global__ void k_deg_init(float* __restrict__ deg) {       // 3*N entries, self-loop = 1
    int i = blockIdx.x * blockDim.x + threadIdx.x;
    if (i < 3 * NNODES) deg[i] = 1.0f;
}

__global__ void k_deg_count(const int* __restrict__ ei, float* __restrict__ deg) {
    int e = blockIdx.x * blockDim.x + threadIdx.x;
    if (e < NEDGES) unsafeAtomicAdd(&deg[ei[(size_t)NEDGES + e]], 1.0f);
}

__global__ void k_rsqrt(float* __restrict__ deg) {
    int i = blockIdx.x * blockDim.x + threadIdx.x;
    if (i < 3 * NNODES) deg[i] = rsqrtf(deg[i]);
}

// ---------------- GEMM1: Htmp[N x 512] = Xbf[N x 256] @ WcatPack ----------------
// 8 waves/block; block handles 16 rows x all 512 cols; wave handles 4 n-tiles.
__global__ void __launch_bounds__(256) k_gemm1(const unsigned short* __restrict__ Xbf,
                                               const unsigned short* __restrict__ Wp,
                                               float* __restrict__ Htmp) {
    const int wave = threadIdx.x >> 5;
    const int lane = threadIdx.x & 31;
    const int row0 = blockIdx.x * 16;
    const int m  = lane & 15;
    const int kh = (lane >> 4) * 8;
    const int KT = FIN >> 5;       // 8
    v8f acc[4] = {};
    const unsigned short* xrow = Xbf + (size_t)(row0 + m) * FIN;
    for (int kt = 0; kt < KT; ++kt) {
        int k0 = kt * 32;
        ABu a;
        a.q[0] = *(const uint4*)(xrow + k0 + 2 * kh);
        a.q[1] = *(const uint4*)(xrow + k0 + 2 * kh + 8);
#pragma unroll
        for (int j = 0; j < 4; ++j) {
            int nt = wave * 4 + j;
            const unsigned short* bp = Wp + (size_t)((nt * KT + kt) * 32 + lane) * 16;
            ABu b;
            b.q[0] = *(const uint4*)(bp);
            b.q[1] = *(const uint4*)(bp + 8);
            acc[j] = __builtin_amdgcn_wmma_f32_16x16x32_bf16(false, a.v, false, b.v,
                                                             (short)0, acc[j], false, false);
        }
    }
#pragma unroll
    for (int j = 0; j < 4; ++j) {
        int n  = (wave * 4 + j) * 16 + (lane & 15);
        int mb = (lane >> 4) * 8;
#pragma unroll
        for (int v = 0; v < 8; ++v)
            Htmp[(size_t)(row0 + mb + v) * FCAT + n] = acc[j][v];
    }
}

// ---------------- Hcat init: bias + self-loop term ----------------
__global__ void k_init_hcat(const float* __restrict__ Htmp, float* __restrict__ Hcat,
                            const float* __restrict__ bln, const float* __restrict__ b1,
                            const float* __restrict__ b2,  const float* __restrict__ b3,
                            const float* __restrict__ dinv) {
    long long i = (long long)blockIdx.x * blockDim.x + threadIdx.x;
    if (i >= (long long)NNODES * FCAT) return;
    int f = (int)(i & (FCAT - 1));
    int node = (int)(i >> 9);
    int c = f >> 7, ff = f & 127;
    float h = Htmp[i];
    float r;
    if (c == 0) {
        r = h + bln[ff];
    } else {
        float di = dinv[(size_t)(c - 1) * NNODES + node];
        const float* bb = (c == 1) ? b1 : (c == 2) ? b2 : b3;
        r = di * di * h + bb[ff];
    }
    Hcat[i] = r;
}

// ---------------- edge aggregation: 32 lanes per edge, float4 per lane ----------------
__global__ void __launch_bounds__(256) k_aggregate(const int* __restrict__ ei,
                                                   const float* __restrict__ dinv,
                                                   const float* __restrict__ Htmp,
                                                   float* __restrict__ Hcat, int coff) {
    int t = threadIdx.x & 31;
    long long e = (long long)blockIdx.x * 8 + (threadIdx.x >> 5);
    if (e >= NEDGES) return;
    int src = ei[e];
    int dst = ei[(size_t)NEDGES + e];
    float w = dinv[src] * dinv[dst];
    const float4 hv = *(const float4*)(Htmp + (size_t)src * FCAT + coff + t * 4);
    float* o = Hcat + (size_t)dst * FCAT + coff + t * 4;
    unsafeAtomicAdd(o + 0, w * hv.x);
    unsafeAtomicAdd(o + 1, w * hv.y);
    unsafeAtomicAdd(o + 2, w * hv.z);
    unsafeAtomicAdd(o + 3, w * hv.w);
}

// ---------------- final GEMM + bias + log_softmax ----------------
// 4 waves/block; each wave one 16-row tile (rows contiguous), all 64 cols; K = 512.
__global__ void __launch_bounds__(128) k_out(const unsigned short* __restrict__ Hbf,
                                             const unsigned short* __restrict__ WoP,
                                             const float* __restrict__ bout,
                                             float* __restrict__ out) {
    __shared__ float lds[64 * CLS];
    const int wave = threadIdx.x >> 5;
    const int lane = threadIdx.x & 31;
    const int tile = blockIdx.x * 4 + wave;
    const bool active = tile < (NNODES / 16);
    if (active) {
        const int row0 = tile * 16;
        const int m  = lane & 15;
        const int kh = (lane >> 4) * 8;
        const int KT = FCAT >> 5;   // 16
        v8f acc[4] = {};
        const unsigned short* hrow = Hbf + (size_t)(row0 + m) * FCAT;
        for (int kt = 0; kt < KT; ++kt) {
            int k0 = kt * 32;
            ABu a;
            a.q[0] = *(const uint4*)(hrow + k0 + 2 * kh);
            a.q[1] = *(const uint4*)(hrow + k0 + 2 * kh + 8);
#pragma unroll
            for (int j = 0; j < 4; ++j) {
                const unsigned short* bp = WoP + (size_t)((j * KT + kt) * 32 + lane) * 16;
                ABu b;
                b.q[0] = *(const uint4*)(bp);
                b.q[1] = *(const uint4*)(bp + 8);
                acc[j] = __builtin_amdgcn_wmma_f32_16x16x32_bf16(false, a.v, false, b.v,
                                                                 (short)0, acc[j], false, false);
            }
        }
#pragma unroll
        for (int j = 0; j < 4; ++j) {
            int n  = j * 16 + (lane & 15);
            int mb = (lane >> 4) * 8;
            float bv = bout[n];
#pragma unroll
            for (int v = 0; v < 8; ++v)
                lds[(wave * 16 + mb + v) * CLS + n] = acc[j][v] + bv;
        }
    }
    __syncthreads();
    int r = threadIdx.x;
    if (r < 64) {
        long long grow = (long long)blockIdx.x * 64 + r;
        if (grow < NNODES) {
            float mx = -3.0e38f;
            for (int c = 0; c < CLS; ++c) mx = fmaxf(mx, lds[r * CLS + c]);
            float s = 0.f;
            for (int c = 0; c < CLS; ++c) s += __expf(lds[r * CLS + c] - mx);
            float lse = mx + __logf(s);
            for (int c = 0; c < CLS; ++c)
                out[grow * CLS + c] = lds[r * CLS + c] - lse;
        }
    }
}

// ---------------- host ----------------

extern "C" void kernel_launch(void* const* d_in, const int* in_sizes, int n_in,
                              void* d_out, int out_size, void* d_ws, size_t ws_size,
                              hipStream_t stream) {
    const float* x    = (const float*)d_in[0];
    const float* W_ln = (const float*)d_in[1];
    const float* b_ln = (const float*)d_in[2];
    const float* W1   = (const float*)d_in[3];
    const float* b1   = (const float*)d_in[4];
    const float* W2   = (const float*)d_in[5];
    const float* b2   = (const float*)d_in[6];
    const float* W3   = (const float*)d_in[7];
    const float* b3   = (const float*)d_in[8];
    const float* Wout = (const float*)d_in[9];
    const float* bout = (const float*)d_in[10];
    const int* ei0 = (const int*)d_in[11];
    const int* ei1 = (const int*)d_in[12];
    const int* ei2 = (const int*)d_in[13];
    float* out = (float*)d_out;

    // workspace carving (256B aligned regions)
    char* ws = (char*)d_ws;
    size_t off = 0;
    auto alloc = [&](size_t bytes) { char* p = ws + off; off = (off + bytes + 255) & ~(size_t)255; return p; };
    unsigned short* Xbf   = (unsigned short*)alloc((size_t)NNODES * FIN * 2);   // 51.2 MB
    unsigned short* WcatP = (unsigned short*)alloc((size_t)FIN * FCAT * 2);     // 0.26 MB
    unsigned short* WoutP = (unsigned short*)alloc((size_t)FCAT * CLS * 2);     // 64 KB
    float*          dinv  = (float*)alloc((size_t)3 * NNODES * 4);              // 1.2 MB
    float*          Htmp  = (float*)alloc((size_t)NNODES * FCAT * 4);           // 204.8 MB
    float*          Hcat  = (float*)alloc((size_t)NNODES * FCAT * 4);           // 204.8 MB
    unsigned short* Hbf   = (unsigned short*)Htmp;  // reuse: Htmp dead after aggregation

    // 1) conversions / packing
    {
        long long n = (long long)NNODES * FIN;
        k_cvt_bf16<<<(unsigned)((n + 255) / 256), 256, 0, stream>>>(x, Xbf, n);
    }
    k_pack_wcat<<<(FIN * FCAT + 255) / 256, 256, 0, stream>>>(W_ln, W1, W2, W3, WcatP);
    k_pack_b<<<(FCAT * CLS + 255) / 256, 256, 0, stream>>>(Wout, WoutP, FCAT, CLS);

    // 2) degrees -> dinv
    k_deg_init<<<(3 * NNODES + 255) / 256, 256, 0, stream>>>(dinv);
    k_deg_count<<<(NEDGES + 255) / 256, 256, 0, stream>>>(ei0, dinv + 0 * NNODES);
    k_deg_count<<<(NEDGES + 255) / 256, 256, 0, stream>>>(ei1, dinv + 1 * NNODES);
    k_deg_count<<<(NEDGES + 255) / 256, 256, 0, stream>>>(ei2, dinv + 2 * NNODES);
    k_rsqrt<<<(3 * NNODES + 255) / 256, 256, 0, stream>>>(dinv);

    // 3) fused GEMM: Htmp = x @ [W_ln|W1|W2|W3]
    k_gemm1<<<NNODES / 16, 256, 0, stream>>>(Xbf, WcatP, Htmp);

    // 4) Hcat init with bias + self-loop term
    {
        long long n = (long long)NNODES * FCAT;
        k_init_hcat<<<(unsigned)((n + 255) / 256), 256, 0, stream>>>(Htmp, Hcat, b_ln, b1, b2, b3, dinv);
    }

    // 5) edge scatter-add (3 convs)
    k_aggregate<<<NEDGES / 8, 256, 0, stream>>>(ei0, dinv + 0 * NNODES, Htmp, Hcat, 1 * HID);
    k_aggregate<<<NEDGES / 8, 256, 0, stream>>>(ei1, dinv + 1 * NNODES, Htmp, Hcat, 2 * HID);
    k_aggregate<<<NEDGES / 8, 256, 0, stream>>>(ei2, dinv + 2 * NNODES, Htmp, Hcat, 3 * HID);

    // 6) convert Hcat -> bf16 (into Htmp's space, now dead)
    {
        long long n = (long long)NNODES * FCAT;
        k_cvt_bf16<<<(unsigned)((n + 255) / 256), 256, 0, stream>>>(Hcat, Hbf, n);
    }

    // 7) classifier GEMM + bias + log_softmax
    k_out<<<(NNODES / 16 + 3) / 4, 128, 0, stream>>>(Hbf, WoutP, bout, out);
    (void)in_sizes; (void)n_in; (void)out_size; (void)ws_size;
}
